// ResonanceEmotionDetector_12489764897512
// MI455X (gfx1250) — matrix operand order repaired and checked by reading, not compiled, not run on hardware
//
#include <hip/hip_runtime.h>
#include <math.h>

typedef __attribute__((ext_vector_type(16))) _Float16 v16h;
typedef __attribute__((ext_vector_type(8)))  _Float16 v8h;
typedef __attribute__((ext_vector_type(8)))  float    v8f;

#define B_   2
#define N_   2048
#define V_   32000
#define D_   1024
#define H_   512
#define ROWS (B_ * N_)      // 4096

// ---------------------------------------------------------------------------
// Kernel 1: online softmax, probability of target token. 524 MB streamed.
// One workgroup (256 threads) per (b,n) row; b128 loads + global_prefetch_b8
// one stride ahead; LDS combine of (max, scaled-sum) pairs.
// ---------------------------------------------------------------------------
__global__ __launch_bounds__(256)
void softmax_target_k(const float* __restrict__ pred,
                      const int*   __restrict__ target,
                      float*       __restrict__ err) {
  const int row = blockIdx.x;
  const int tid = threadIdx.x;
  const float* base = pred + (size_t)row * V_;

  float m = -INFINITY, s = 0.0f;
  const float4* b4 = (const float4*)base;
  const int nv = V_ / 4;                       // 8000, divisible
  for (int i = tid; i < nv; i += 256) {
    if (i + 256 < nv) __builtin_prefetch(&b4[i + 256], 0, 3); // global_prefetch_b8
    float4 v = b4[i];
    float x[4] = {v.x, v.y, v.z, v.w};
#pragma unroll
    for (int j = 0; j < 4; ++j) {
      float nm = fmaxf(m, x[j]);
      s = s * __expf(m - nm) + __expf(x[j] - nm);
      m = nm;
    }
  }

  __shared__ float sm[256], ss[256];
  sm[tid] = m; ss[tid] = s;
  __syncthreads();
  for (int off = 128; off > 0; off >>= 1) {
    if (tid < off) {
      float m2 = sm[tid + off], s2 = ss[tid + off];
      float nm = fmaxf(sm[tid], m2);
      ss[tid] = ss[tid] * __expf(sm[tid] - nm) + s2 * __expf(m2 - nm);
      sm[tid] = nm;
    }
    __syncthreads();
  }
  if (tid == 0) {
    float xt = base[target[row]];
    float p  = __expf(xt - sm[0]) / ss[0];
    err[row] = 1.0f - p;
  }
}

// ---------------------------------------------------------------------------
// Kernel 1b: one-time f32 -> f16 conversion of X (row-major) and W1
// (transposed to column-major) so the WMMA loop can use pure b128 loads.
// ---------------------------------------------------------------------------
__global__ __launch_bounds__(256)
void convert_x_k(const float* __restrict__ X, _Float16* __restrict__ Xh) {
  const int i = blockIdx.x * 256 + threadIdx.x;   // over ROWS*D_/4
  const float4 v = ((const float4*)X)[i];
  v8h* dst = (v8h*)Xh;
  // pack 4 halves; write as 64-bit chunk
  _Float16 h4[4] = {(_Float16)v.x, (_Float16)v.y, (_Float16)v.z, (_Float16)v.w};
  ((unsigned long long*)dst)[i] = *(unsigned long long*)h4;
}

__global__ __launch_bounds__(256)
void convert_w1_k(const float* __restrict__ W1, _Float16* __restrict__ W1t) {
  // W1 is (D_, H_) row-major; W1t is (H_, D_): W1t[n*D_ + k] = W1[k*H_ + n]
  const int i = blockIdx.x * 256 + threadIdx.x;   // over D_*H_
  const int k = i / H_, n = i % H_;
  W1t[(size_t)n * D_ + k] = (_Float16)W1[i];
}

// ---------------------------------------------------------------------------
// Kernel 2: H = tanh(X @ W1 + b1) via v_wmma_f32_16x16x32_f16.
// One wave per 16x16 output tile; K swept in steps of 32.
// A (16x32 f16): lane L (hi=L>>4, ml=L&15) holds row M=ml,
//   elems 0..7  -> K = hi*8 + e        (contiguous 8 halves)
//   elems 8..15 -> K = hi*8 + 16 + e-8 (contiguous 8 halves)
// B (32x16 f16): lane L holds col N=ml, same K order as A:
//   elems 0..7 -> K = hi*8 + e, elems 8..15 -> K = hi*8 + 16 + e-8 (column-major W1t).
// C/D (16x16 f32): vgpr r, lane L -> M = r + 8*hi, N = ml.
// Inner loop: 4x global_load_b128 + 1x v_wmma, zero converts.
// ---------------------------------------------------------------------------
__global__ __launch_bounds__(256)
void mlp1_wmma_k(const _Float16* __restrict__ Xh,
                 const _Float16* __restrict__ W1t,
                 const float*    __restrict__ b1,
                 float*          __restrict__ H) {
  const int wave = (blockIdx.x * blockDim.x + threadIdx.x) >> 5;
  const int lane = threadIdx.x & 31;
  const int tn   = wave & 31;          // 512/16 = 32 tiles along N
  const int tm   = wave >> 5;          // 4096/16 = 256 tiles along M
  const int m0 = tm * 16, n0 = tn * 16;
  const int hi = lane >> 4, ml = lane & 15;

  v8f acc = {};
  const _Float16* xrow = Xh  + (size_t)(m0 + ml) * D_;
  const _Float16* wcol = W1t + (size_t)(n0 + ml) * D_;

#pragma unroll 2
  for (int kb = 0; kb < D_; kb += 32) {
    v8h a0 = *(const v8h*)(xrow + kb + hi * 8);
    v8h a1 = *(const v8h*)(xrow + kb + hi * 8 + 16);
    v8h b0 = *(const v8h*)(wcol + kb + hi * 8);
    v8h b1 = *(const v8h*)(wcol + kb + hi * 8 + 16);
    v16h a = __builtin_shufflevector(a0, a1, 0,1,2,3,4,5,6,7,8,9,10,11,12,13,14,15);
    v16h b = __builtin_shufflevector(b0, b1, 0,1,2,3,4,5,6,7,8,9,10,11,12,13,14,15);
    acc = __builtin_amdgcn_wmma_f32_16x16x32_f16(false, a, false, b,
                                                 (short)0, acc, false, false);
  }

  const float bias = b1[n0 + ml];
#pragma unroll
  for (int r = 0; r < 8; ++r) {
    const int mrow = m0 + r + hi * 8;
    H[(size_t)mrow * H_ + n0 + ml] = tanhf(acc[r] + bias);
  }
}

// ---------------------------------------------------------------------------
// Kernel 3: sigma = softplus(H @ W2 + b2); D = sqrt(err*sigma + 1e-9).
// One workgroup (128 threads) per row.
// ---------------------------------------------------------------------------
__global__ __launch_bounds__(128)
void sigma_d_k(const float* __restrict__ H,
               const float* __restrict__ W2,
               const float* __restrict__ b2,
               const float* __restrict__ err,
               float*       __restrict__ Dm) {
  const int row = blockIdx.x;
  const int tid = threadIdx.x;
  float p = 0.0f;
  for (int j = tid; j < H_; j += 128)
    p += H[(size_t)row * H_ + j] * W2[j];
  __shared__ float red[128];
  red[tid] = p;
  __syncthreads();
  for (int off = 64; off > 0; off >>= 1) {
    if (tid < off) red[tid] += red[tid + off];
    __syncthreads();
  }
  if (tid == 0) {
    float z  = red[0] + b2[0];
    float sp = (z > 20.0f) ? z : log1pf(__expf(z));
    float v  = err[row] * sp;
    Dm[row]  = sqrtf(v + 1e-9f);
  }
}

// ---------------------------------------------------------------------------
// Kernel 4: LDS-resident power iteration. One workgroup per batch (B_=2).
// alpha = exp(i*i) = e^-1 (real). Bidirectional IIR via chunked parallel
// scan: 256 threads x 8-element serial chunks, carry chain weighted by
// alpha^8 (~3.3e-4), serial 256-step carry prefix on thread 0.
// ---------------------------------------------------------------------------
#define PI_T  256
#define CHUNK 8

__device__ __forceinline__ unsigned int hash_u32(unsigned int x) {
  x ^= x >> 16; x *= 0x7feb352du;
  x ^= x >> 15; x *= 0x846ca68bu;
  x ^= x >> 16;
  return x;
}

__device__ __forceinline__ float block_sum(float v, float* red, int tid) {
  red[tid] = v;
  __syncthreads();
  for (int off = PI_T / 2; off > 0; off >>= 1) {
    if (tid < off) red[tid] += red[tid + off];
    __syncthreads();
  }
  float r = red[0];
  __syncthreads();
  return r;
}

__global__ __launch_bounds__(PI_T)
void power_iter_k(const float* __restrict__ Dm, float* __restrict__ out) {
  const int batch = blockIdx.x;
  const int tid   = threadIdx.x;

  __shared__ float2 sb[N_];        // b_k
  __shared__ float2 su[N_];        // u = D*b, later w
  __shared__ float2 sf[N_];        // forward scan
  __shared__ float  sd[N_];        // D (real)
  __shared__ float2 carr[PI_T];    // scan carries / complex reduction
  __shared__ float  red[PI_T];

  const float AL = 0.36787944117144233f;      // e^-1
  const float ALp[CHUNK] = {                  // alpha^(j+1), j=0..7
      0.36787944f, 0.13533528f, 0.049787068f, 0.018315639f,
      0.0067379470f, 0.0024787522f, 0.00091188197f, 0.00033546263f};
  const float ALC = ALp[CHUNK - 1];           // alpha^CHUNK

  // ---- load D, deterministic gaussian init for b ----
  for (int i = tid; i < N_; i += PI_T) {
    sd[i] = Dm[batch * N_ + i];
    unsigned int h1 = hash_u32(0x9E3779B9u ^ (unsigned)(batch * N_ + i));
    unsigned int h2 = hash_u32(h1 ^ 0x85EBCA6Bu);
    unsigned int h3 = hash_u32(h2 ^ 0xC2B2AE35u);
    unsigned int h4 = hash_u32(h3 + 0x27D4EB2Fu);
    float u1 = ((float)h1 + 0.5f) * 2.3283064e-10f;
    float u2 = ((float)h2 + 0.5f) * 2.3283064e-10f;
    float u3 = ((float)h3 + 0.5f) * 2.3283064e-10f;
    float u4 = ((float)h4 + 0.5f) * 2.3283064e-10f;
    float r1 = sqrtf(-2.0f * __logf(u1));
    float r2 = sqrtf(-2.0f * __logf(u3));
    sb[i] = make_float2(r1 * __cosf(6.2831853f * u2),
                        r2 * __cosf(6.2831853f * u4));
  }
  __syncthreads();

  // ---- normalize initial b ----
  {
    float p = 0.0f;
    for (int i = tid; i < N_; i += PI_T) p += sb[i].x * sb[i].x + sb[i].y * sb[i].y;
    float inv = 1.0f / fmaxf(sqrtf(block_sum(p, red, tid)), 1e-12f);
    for (int i = tid; i < N_; i += PI_T) { sb[i].x *= inv; sb[i].y *= inv; }
    __syncthreads();
  }

  const int base = tid * CHUNK;
  float2 eig = make_float2(0.0f, 0.0f);

  for (int it = 0; it <= 15; ++it) {
    // u = D * b
    for (int i = tid; i < N_; i += PI_T)
      su[i] = make_float2(sd[i] * sb[i].x, sd[i] * sb[i].y);
    __syncthreads();

    // ---- forward scan: fwd[t] = u[t] + AL*fwd[t-1] ----
    {
      float2 loc[CHUNK];
      float2 c = make_float2(0.0f, 0.0f);
#pragma unroll
      for (int j = 0; j < CHUNK; ++j) {
        float2 v = su[base + j];
        c = make_float2(v.x + AL * c.x, v.y + AL * c.y);
        loc[j] = c;
      }
      carr[tid] = c;
      __syncthreads();
      if (tid == 0) {
        float2 pc = make_float2(0.0f, 0.0f);
        for (int t = 0; t < PI_T; ++t) {
          float2 cl = carr[t];
          carr[t] = pc;
          pc = make_float2(cl.x + ALC * pc.x, cl.y + ALC * pc.y);
        }
      }
      __syncthreads();
      float2 pc = carr[tid];
#pragma unroll
      for (int j = 0; j < CHUNK; ++j)
        sf[base + j] = make_float2(loc[j].x + ALp[j] * pc.x,
                                   loc[j].y + ALp[j] * pc.y);
      __syncthreads();
    }

    // ---- backward scan + combine: w = D * 0.5i*(fwd + bwd - u) -> su ----
    {
      float2 loc[CHUNK];
      float2 c = make_float2(0.0f, 0.0f);
#pragma unroll
      for (int j = CHUNK - 1; j >= 0; --j) {
        float2 v = su[base + j];
        c = make_float2(v.x + AL * c.x, v.y + AL * c.y);
        loc[j] = c;
      }
      carr[tid] = c;
      __syncthreads();
      if (tid == 0) {
        float2 pc = make_float2(0.0f, 0.0f);
        for (int t = PI_T - 1; t >= 0; --t) {
          float2 cl = carr[t];
          carr[t] = pc;
          pc = make_float2(cl.x + ALC * pc.x, cl.y + ALC * pc.y);
        }
      }
      __syncthreads();
      float2 pc = carr[tid];
#pragma unroll
      for (int j = 0; j < CHUNK; ++j) {
        const int i = base + j;
        float2 bw = make_float2(loc[j].x + ALp[CHUNK - 1 - j] * pc.x,
                                loc[j].y + ALp[CHUNK - 1 - j] * pc.y);
        float2 u  = su[i];
        float2 t  = make_float2(sf[i].x + bw.x - u.x, sf[i].y + bw.y - u.y);
        // 0.5i * t = (-0.5*t.y, 0.5*t.x); then scale by real D
        su[i] = make_float2(sd[i] * (-0.5f * t.y), sd[i] * (0.5f * t.x));
      }
      __syncthreads();
    }

    if (it < 15) {
      // b = normalize(w)
      float p = 0.0f;
      for (int i = tid; i < N_; i += PI_T) p += su[i].x * su[i].x + su[i].y * su[i].y;
      float inv = 1.0f / fmaxf(sqrtf(block_sum(p, red, tid)), 1e-12f);
      for (int i = tid; i < N_; i += PI_T)
        sb[i] = make_float2(su[i].x * inv, su[i].y * inv);
      __syncthreads();
    } else {
      // eigenvalue = sum(conj(b) * w)
      float2 p = make_float2(0.0f, 0.0f);
      for (int i = tid; i < N_; i += PI_T) {
        float2 b = sb[i], w = su[i];
        p.x += b.x * w.x + b.y * w.y;
        p.y += b.x * w.y - b.y * w.x;
      }
      carr[tid] = p;
      __syncthreads();
      for (int off = PI_T / 2; off > 0; off >>= 1) {
        if (tid < off) {
          carr[tid].x += carr[tid + off].x;
          carr[tid].y += carr[tid + off].y;
        }
        __syncthreads();
      }
      eig = carr[0];
    }
  }

  if (tid == 0) {
    const float PI_F = 3.14159265358979f;
    float phase  = atan2f(eig.y, eig.x);
    float interf = fabsf(eig.y);
    out[batch]           = interf * __cosf(phase);                 // resonance
    out[2 + batch]       = interf * fabsf(phase - PI_F) / PI_F;    // dissonance
    out[4 + 2 * batch]   = eig.x;                                  // dom_eig.re
    out[5 + 2 * batch]   = eig.y;                                  // dom_eig.im
  }
}

// ---------------------------------------------------------------------------
extern "C" void kernel_launch(void* const* d_in, const int* in_sizes, int n_in,
                              void* d_out, int out_size, void* d_ws, size_t ws_size,
                              hipStream_t stream) {
  const float* pred   = (const float*)d_in[0];   // (2, 2048, 32000)
  const int*   target = (const int*)  d_in[1];   // (2, 2048)
  const float* X      = (const float*)d_in[2];   // (2, 2048, 1024)
  const float* W1     = (const float*)d_in[3];   // (1024, 512)
  const float* b1     = (const float*)d_in[4];   // (512,)
  const float* W2     = (const float*)d_in[5];   // (512, 1)
  const float* b2     = (const float*)d_in[6];   // (1,)
  float* out = (float*)d_out;

  float*    err = (float*)d_ws;                    // ROWS floats
  float*    Dm  = err + ROWS;                      // ROWS floats
  float*    H   = Dm  + ROWS;                      // ROWS*H_ floats (8 MB)
  _Float16* Xh  = (_Float16*)(H + (size_t)ROWS * H_);       // ROWS*D_ halves (8 MB)
  _Float16* W1t = Xh + (size_t)ROWS * D_;                   // H_*D_ halves (1 MB)

  // 1) streaming online softmax (bandwidth-bound, ~524 MB -> ~22.5 us floor)
  softmax_target_k<<<ROWS, 256, 0, stream>>>(pred, target, err);

  // 1b) one-time f16 conversion / transpose for WMMA-friendly b128 loads
  convert_x_k<<<(ROWS * D_ / 4) / 256, 256, 0, stream>>>(X, Xh);
  convert_w1_k<<<(D_ * H_) / 256, 256, 0, stream>>>(W1, W1t);

  // 2) WMMA MLP layer 1: 8192 waves, one 16x16 tile each
  const int waves  = (ROWS / 16) * (H_ / 16);   // 8192
  const int blocks = waves * 32 / 256;          // 1024
  mlp1_wmma_k<<<blocks, 256, 0, stream>>>(Xh, W1t, b1, H);

  // 3) layer 2 + softplus + D
  sigma_d_k<<<ROWS, 128, 0, stream>>>(H, W2, b2, err, Dm);

  // 4) LDS-resident power iteration, one workgroup per batch
  power_iter_k<<<B_, PI_T, 0, stream>>>(Dm, out);
}
